// MultiHeadAttention_47777216201262
// MI455X (gfx1250) — matrix operand order repaired and checked by reading, not compiled, not run on hardware
//
#include <hip/hip_runtime.h>
#include <hip/hip_bf16.h>

// B=4, S=2048, D=1024, H=16, DK=64
#define BB 4
#define SS 2048
#define DD 1024
#define HH 16
#define DKK 64
#define KVP 68   // padded LDS row stride (ushorts) for the V tile: bank-shifts lane halves

typedef __attribute__((ext_vector_type(16))) __bf16         v16bf;
typedef __attribute__((ext_vector_type(8)))  float          v8f;
typedef __attribute__((ext_vector_type(16))) unsigned short v16us;
typedef __attribute__((ext_vector_type(8)))  unsigned short v8us;
typedef __attribute__((ext_vector_type(4)))  unsigned int   v4u;
typedef __attribute__((ext_vector_type(8)))  int            v8i;
typedef __attribute__((ext_vector_type(4)))  int            v4i;

__device__ __forceinline__ unsigned short f2bf(float f) {
    unsigned int u = __builtin_bit_cast(unsigned int, f);
    u += 0x7fffu + ((u >> 16) & 1u);            // round to nearest even
    return (unsigned short)(u >> 16);
}

// ---------------------------------------------------------------------------
// Kernel 0: fp32 x (B,S,D) -> bf16 xbf (B,H,S,DK); zero csum (B*D)
// ---------------------------------------------------------------------------
__global__ void prep_kernel(const float* __restrict__ x,
                            unsigned short* __restrict__ xbf,
                            float* __restrict__ csum) {
    int idx = blockIdx.x * 256 + threadIdx.x;           // B*H*S*DK = 8388608
    int d = idx & 63;
    int s = (idx >> 6) & 2047;
    int h = (idx >> 17) & 15;
    int b = idx >> 21;
    int in = (b * SS + s) * DD + h * DKK + d;
    xbf[idx] = f2bf(x[in]);
    if (idx < BB * DD) csum[idx] = 0.0f;
}

// ---------------------------------------------------------------------------
// Kernel 1: per (b,h, 16-row q tile):
//   scores (WMMA bf16, fp32 acc) -> masked softmax (fp32, LDS)
//   -> async TDM store of the attn tile (overlaps with ctx compute)
//   -> ctx = attn @ V (WMMA bf16, V staged coalesced into LDS)
//   -> atomic-accumulate sum_s ctx into csum[b][h*64+dk]
// grid = (S/16, B*H), block = 256 (8 wave32)
// ---------------------------------------------------------------------------
__global__ void attn_kernel(const unsigned short* __restrict__ xbf,
                            const long long* __restrict__ seq_lens,
                            float* __restrict__ attn_out,
                            float* __restrict__ csum) {
    extern __shared__ float smem[];
    float* sc     = smem;                 // 16*2048 floats, MUST be at LDS offset 0 (TDM)
    float* ctxsum = smem + 16 * SS;       // 16*64
    float* red    = ctxsum + 1024;        // 256
    float* rowM   = red + 256;            // 16
    float* rowS   = rowM + 16;            // 16
    unsigned short* kv = (unsigned short*)(rowS + 16); // 8 waves * 32 * KVP ushorts

    const int tid  = threadIdx.x;
    const int lane = tid & 31;
    const int wave = tid >> 5;
    const int lhi  = (lane >= 16) ? 1 : 0;
    const int l15  = lane & 15;

    const int q0 = blockIdx.x * 16;
    const int bh = blockIdx.y;
    const int b  = bh >> 4;
    const int h  = bh & 15;
    const int slen = (int)seq_lens[b];

    const unsigned short* xh = xbf + (size_t)bh * SS * DKK;

    for (int i = tid; i < 1024; i += 256) ctxsum[i] = 0.0f;

    // ---- A operand: q tile, 16(M) x 64(dk), contiguous k = 16h + s ----------
    const unsigned short* qp = xh + (q0 + l15) * DKK;
    const int ao = lhi ? 16 : 0;
    v8us qa0 = *(const v8us*)(qp + ao);
    v8us qb0 = *(const v8us*)(qp + ao + 8);
    v8us qa1 = *(const v8us*)(qp + ao + 32);
    v8us qb1 = *(const v8us*)(qp + ao + 40);
    v16bf a_lo = __builtin_bit_cast(v16bf, __builtin_shufflevector(
        qa0, qb0, 0,1,2,3,4,5,6,7,8,9,10,11,12,13,14,15));
    v16bf a_hi = __builtin_bit_cast(v16bf, __builtin_shufflevector(
        qa1, qb1, 0,1,2,3,4,5,6,7,8,9,10,11,12,13,14,15));

    // ---- scores: each wave owns 256 keys = 16 tiles of 16 ------------------
    const int kbase = wave * 256;
    const int mbase = lhi ? 8 : 0;
    for (int kt = 0; kt < 16; ++kt) {
        int k0  = kbase + kt * 16;
        int key = k0 + l15;
        const unsigned short* kp = xh + key * DKK + (lhi ? 16 : 0);
        __builtin_prefetch((const void*)(kp + 16 * DKK), 0, 0); // next key tile
        v16bf b_lo = __builtin_bit_cast(v16bf, *(const v16us*)(kp));
        v16bf b_hi = __builtin_bit_cast(v16bf, *(const v16us*)(kp + 32));
        v8f c = {};
        c = __builtin_amdgcn_wmma_f32_16x16x32_bf16(false, a_lo, false, b_lo,
                                                    (short)0, c, false, false);
        c = __builtin_amdgcn_wmma_f32_16x16x32_bf16(false, a_hi, false, b_hi,
                                                    (short)0, c, false, false);
        int col = k0 + l15;
        bool valid = col < slen;
#pragma unroll
        for (int r = 0; r < 8; ++r) {
            float v = valid ? (c[r] * 0.125f) : -1.0e9f;
            sc[(mbase + r) * SS + col] = v;
        }
    }
    __syncthreads();

    // ---- softmax over 2048 keys: 16 threads per row ------------------------
    const int row  = tid >> 4;
    const int part = tid & 15;
    float m = -3.0e38f;
    for (int j = 0; j < 128; ++j)
        m = fmaxf(m, sc[row * SS + part + 16 * j]);
    red[tid] = m;
    __syncthreads();
    if (tid < 16) {
        float mm = red[tid * 16];
        for (int i = 1; i < 16; ++i) mm = fmaxf(mm, red[tid * 16 + i]);
        rowM[tid] = mm;
    }
    __syncthreads();
    float rm = rowM[row], s = 0.0f;
    for (int j = 0; j < 128; ++j) {
        int o = row * SS + part + 16 * j;
        float e = __expf(sc[o] - rm);
        sc[o] = e;
        s += e;
    }
    red[tid] = s;
    __syncthreads();
    if (tid < 16) {
        float ss = 0.0f;
        for (int i = 0; i < 16; ++i) ss += red[tid * 16 + i];
        rowS[tid] = ss;
    }
    __syncthreads();
    float inv = 1.0f / rowS[row];
    for (int j = 0; j < 128; ++j)
        sc[row * SS + part + 16 * j] *= inv;
    __syncthreads();

    // ---- async TDM store of the normalized 16x2048 fp32 tile ---------------
    // D# per CDNA5 ISA; tile = full sc buffer at LDS offset 0.
    if (wave == 0) {
        size_t abase = ((size_t)bh * SS + q0) * SS;
        unsigned long long ga =
            (unsigned long long)(uintptr_t)(attn_out + abase);
        v4u g0;
        g0[0] = 1u;                                   // count=1, user mode
        g0[1] = 0u;                                   // lds_addr = 0 (sc base)
        g0[2] = (unsigned int)ga;                     // global_addr[31:0]
        g0[3] = (unsigned int)((ga >> 32) & 0x01FFFFFFull) | 0x80000000u; // addr[56:32] | type=2
        v8i g1;
        g1[0] = 0x00020000;                           // data_size = 4B
        g1[1] = 0x08000000;                           // tensor_dim0 = 2048 (bits 79:48)
        g1[2] = 0x00100000;                           // tensor_dim1 = 16   (bits 111:80)
        g1[3] = 0x08000000;                           // tile_dim0 = 2048   (bits 127:112)
        g1[4] = 0x00000010;                           // tile_dim1 = 16
        g1[5] = 2048;                                 // tensor_dim0_stride
        g1[6] = 0;
        g1[7] = 0;
        v4i gz = {0, 0, 0, 0};
#if defined(__clang_major__) && (__clang_major__ >= 23)
        v8i gz8 = {0, 0, 0, 0, 0, 0, 0, 0};
        __builtin_amdgcn_tensor_store_from_lds(g0, g1, gz, gz, gz8, 0);
#else
        __builtin_amdgcn_tensor_store_from_lds(g0, g1, gz, gz, 0);
#endif
    }

    // ---- ctx = attn(16x2048) @ V(2048x64); wave handles its 256 keys -------
    v8f acc[4] = {{}, {}, {}, {}};
    const int koffA = lhi ? 16 : 0;
    const int koffB = lhi ? 16 : 0;
    unsigned short* kvw = kv + wave * 32 * KVP;
    for (int ks = 0; ks < 8; ++ks) {
        int kk0 = kbase + ks * 32;

        // stage 32 keys x 64 dk into LDS, coalesced 16B chunks
        const unsigned short* src = xh + (size_t)kk0 * DKK;
#pragma unroll
        for (int c = 0; c < 8; ++c) {
            int ci  = c * 32 + lane;          // chunk index 0..255
            int r   = ci >> 3;                // key row
            int col = (ci & 7) * 8;           // dk start
            v4u d = __builtin_bit_cast(v4u, *(const v8us*)(src + ci * 8));
            unsigned int* dst = (unsigned int*)(kvw + r * KVP + col);
            dst[0] = d[0]; dst[1] = d[1]; dst[2] = d[2]; dst[3] = d[3];
        }

        // A operand: attn rows from sc (fp32 -> bf16)
        v16us au;
#pragma unroll
        for (int e = 0; e < 8; ++e)
            au[e] = f2bf(sc[l15 * SS + kk0 + koffA + e]);
#pragma unroll
        for (int e = 0; e < 8; ++e)
            au[8 + e] = f2bf(sc[l15 * SS + kk0 + koffA + 8 + e]);
        v16bf aA = __builtin_bit_cast(v16bf, au);

#pragma unroll
        for (int dkt = 0; dkt < 4; ++dkt) {
            int dk = dkt * 16 + l15;
            v16us bu;
#pragma unroll
            for (int e = 0; e < 16; ++e)
                bu[e] = kvw[(koffB + e) * KVP + dk];  // LDS gather, bank-shifted
            v16bf bB = __builtin_bit_cast(v16bf, bu);
            acc[dkt] = __builtin_amdgcn_wmma_f32_16x16x32_bf16(
                false, aA, false, bB, (short)0, acc[dkt], false, false);
        }
    }
    // cross-wave reduce partial 16x64 tiles via LDS float atomics
#pragma unroll
    for (int dkt = 0; dkt < 4; ++dkt)
#pragma unroll
        for (int r = 0; r < 8; ++r)
            atomicAdd(&ctxsum[(mbase + r) * 64 + dkt * 16 + l15], acc[dkt][r]);
    __syncthreads();

    // sum over the 16 q rows, accumulate into global csum[b][h*64 + dk]
    if (tid < 64) {
        float ssum = 0.0f;
        for (int mr = 0; mr < 16; ++mr) ssum += ctxsum[mr * 64 + tid];
        atomicAdd(&csum[(size_t)b * DD + h * DKK + tid], ssum);
    }

    __builtin_amdgcn_s_wait_tensorcnt(0);   // TDM store complete before exit
}

// ---------------------------------------------------------------------------
// Kernel 2: out_sum[b][d] = dot(csum[b], W[d,:]) + S * bias[d]
// ---------------------------------------------------------------------------
__global__ void proj_kernel(const float* __restrict__ csum,
                            const float* __restrict__ W,
                            const float* __restrict__ bias,
                            float* __restrict__ out) {
    int idx = blockIdx.x * 256 + threadIdx.x;   // B*D = 4096
    int b = idx >> 10;
    int d = idx & 1023;
    float acc = (float)SS * bias[d];
    const float* wr = W + (size_t)d * DD;
    const float* cs = csum + (size_t)b * DD;
    for (int k = 0; k < DD; ++k) acc += cs[k] * wr[k];
    out[idx] = acc;
}

// ---------------------------------------------------------------------------
extern "C" void kernel_launch(void* const* d_in, const int* in_sizes, int n_in,
                              void* d_out, int out_size, void* d_ws, size_t ws_size,
                              hipStream_t stream) {
    const float*      x   = (const float*)d_in[0];
    const long long*  sl  = (const long long*)d_in[1];
    const float*      W   = (const float*)d_in[2];
    const float*      bia = (const float*)d_in[3];

    // workspace layout: xbf (bf16, B*H*S*DK = 16 MB) | csum (B*D fp32)
    unsigned short* xbf  = (unsigned short*)d_ws;
    float*          csum = (float*)((char*)d_ws + (size_t)BB * HH * SS * DKK * 2);

    float* out_sum  = (float*)d_out;                     // B*D
    float* attn_out = (float*)d_out + (size_t)BB * DD;   // B*H*S*S

    prep_kernel<<<(BB * HH * SS * DKK) / 256, 256, 0, stream>>>(x, xbf, csum);

    size_t lds = (size_t)(16 * SS + 1024 + 256 + 16 + 16) * sizeof(float)
               + (size_t)8 * 32 * KVP * sizeof(unsigned short);
    attn_kernel<<<dim3(SS / 16, BB * HH), 256, lds, stream>>>(xbf, sl, attn_out, csum);

    proj_kernel<<<(BB * DD) / 256, 256, 0, stream>>>(csum, W, bia, out_sum);
}
